// GroupedQueryAttention_18013092839969
// MI455X (gfx1250) — matrix operand order repaired and checked by reading, not compiled, not run on hardware
//
#include <hip/hip_runtime.h>
#include <hip/hip_bf16.h>

typedef __attribute__((ext_vector_type(16))) __bf16 v16bf;
typedef __attribute__((ext_vector_type(8)))  __bf16 v8bf;
typedef __attribute__((ext_vector_type(8)))  float  v8f;
typedef unsigned short u16;

union Frag16 { v16bf v; v8bf h[2]; u16 u[16]; };

__device__ __forceinline__ u16 f2bf(float f) {
  unsigned u = __float_as_uint(f);
  u += 0x7FFFu + ((u >> 16) & 1u);   // round-to-nearest-even
  return (u16)(u >> 16);
}

__device__ __forceinline__ float rmax16(float x) {
  x = fmaxf(x, __shfl_xor(x, 1, 32));
  x = fmaxf(x, __shfl_xor(x, 2, 32));
  x = fmaxf(x, __shfl_xor(x, 4, 32));
  x = fmaxf(x, __shfl_xor(x, 8, 32));
  return x;
}
__device__ __forceinline__ float rsum16(float x) {
  x += __shfl_xor(x, 1, 32);
  x += __shfl_xor(x, 2, 32);
  x += __shfl_xor(x, 4, 32);
  x += __shfl_xor(x, 8, 32);
  return x;
}

// One async 16B-per-lane global->LDS copy (CDNA5; ASYNCcnt-tracked).
__device__ __forceinline__ void async_16B(const u16* g, unsigned ldsb) {
  asm volatile("global_load_async_to_lds_b128 %0, %1, off"
               :: "v"(ldsb), "v"((unsigned long long)(size_t)g)
               : "memory");
}

// ---- elementwise f32 -> bf16 cast ----
__global__ void cast_bf(const float* __restrict__ X, u16* __restrict__ Y, int n) {
  int i = blockIdx.x * 256 + threadIdx.x;
  if (i < n) Y[i] = f2bf(X[i]);
}

// ---- W (K x N) f32 -> Wt (N x K) bf16 ----
__global__ void transpose_w(const float* __restrict__ W, u16* __restrict__ Wt, int K, int N) {
  size_t i = (size_t)blockIdx.x * 256 + threadIdx.x;
  if (i >= (size_t)K * N) return;
  int n = (int)(i / K);
  int k = (int)(i % K);
  Wt[i] = f2bf(W[(size_t)k * N + n]);
}

// ---- V (B,S,512) bf16 -> Vt (B,8,64,S) bf16 ----
__global__ void transpose_v(const u16* __restrict__ V, u16* __restrict__ Vt) {
  const int S = 2048;
  size_t i = (size_t)blockIdx.x * 256 + threadIdx.x;   // over B*8*64*S = 2^21
  int t = (int)(i % S);  size_t rest = i / S;
  int d = (int)(rest % 64); rest /= 64;
  int g = (int)(rest % 8);  int b = (int)(rest / 8);
  Vt[i] = V[((size_t)b * S + t) * 512 + g * 64 + d];
}

// A-fragment: ISA 16-bit A 16x32 layout: lanes 0-15 hold K{0..7,16..23},
// lanes 16-31 hold K{8..15,24..31}; lane l = row (l&15).
__device__ __forceinline__ void load_a(Frag16& a, const u16* row, int k0, int hi) {
  a.h[0] = *(const v8bf*)(row + k0 + hi * 8);
  a.h[1] = *(const v8bf*)(row + k0 + 16 + hi * 8);
}
// B-fragment: col-major source row (= one B column); lanes 0-15 K{0..7,16..23},
// lanes 16-31 K{8..15,24..31} (same order as A).
__device__ __forceinline__ void load_b(Frag16& b, const u16* colrow, int k0, int hi) {
  b.h[0] = *(const v8bf*)(colrow + k0 + hi * 8);
  b.h[1] = *(const v8bf*)(colrow + k0 + 16 + hi * 8);
}

// ---- projection GEMM: Y(S x N) = Xb(S x 2048) @ W + bias, per batch ----
// Wave computes a 16x64 tile; the 64x32 Wt tile is async-staged in LDS and
// shared by all 8 waves (8x L2-traffic cut), double-buffered branchlessly.
__global__ void proj_gemm(const u16* __restrict__ Xb, const u16* __restrict__ Wt,
                          const float* __restrict__ bias, u16* __restrict__ Y, int N) {
  const int S = 2048, K = 2048;
  __shared__ __align__(32) u16 wtile[2][64][32];     // [buf][col][k] 4KB each

  const int b    = blockIdx.z;
  const int tid  = threadIdx.x;
  const int lane = tid & 31;
  const int wave = tid >> 5;
  const int r  = lane & 15;
  const int hi = lane >> 4;
  const int s0 = blockIdx.y * 128 + wave * 16;
  const int n0 = blockIdx.x * 64;

  const u16* arow = Xb + ((size_t)b * S + (s0 + r)) * K;

  // per-thread 16B staging slice: elem offset tid*8 -> col=tid/4, koff=(tid&3)*8
  const u16* wsrc = Wt + (size_t)(n0 + (tid >> 2)) * K + (tid & 3) * 8;
  const unsigned wl0 = (unsigned)(size_t)(&wtile[0][0][0]) + tid * 16u;
  const u16* wt0 = &wtile[0][0][0];

  async_16B(wsrc, wl0);                              // prologue: k0=0 -> buf0

  v8f acc[4] = {};
#pragma unroll 1
  for (int it = 0; it < K / 32; ++it) {
    const int k0 = it * 32;
    const unsigned par = (unsigned)it & 1u;          // current buffer parity
    __syncthreads();                                 // prev readers of other buf done
    const int kn = (k0 + 32 < K) ? k0 + 32 : 0;      // uniform dummy on last iter
    async_16B(wsrc + kn, wl0 + ((par ^ 1u) << 12));  // fill other buf
    asm volatile("s_wait_asynccnt 0x1" ::: "memory"); // my buf-copy landed
    __syncthreads();                                 // everyone's buf-copy visible

    const u16* bt = wt0 + (par << 11);               // 2048 u16 per buffer
    Frag16 a; load_a(a, arow, k0, hi);               // A direct from global
#pragma unroll
    for (int nt = 0; nt < 4; ++nt) {
      Frag16 bf;
      load_b(bf, bt + (nt * 16 + r) * 32, 0, hi);
      acc[nt] = __builtin_amdgcn_wmma_f32_16x16x32_bf16(
          false, a.v, false, bf.v, (short)0, acc[nt], false, false);
    }
  }
#pragma unroll
  for (int nt = 0; nt < 4; ++nt) {
    int n = n0 + nt * 16 + r;
    float bb = bias[n];
#pragma unroll
    for (int j = 0; j < 8; ++j) {
      int m = j + 8 * hi;                 // C layout: lane l, vgpr j -> M=j+8*(l>=16), N=l&15
      Y[((size_t)b * S + (s0 + m)) * N + n] = f2bf(acc[nt][j] + bb);
    }
  }
}

// ---- flash attention: 8 waves x 16 q-rows per WG; K/V tiles async-staged ----
__global__ void gqa_attn(const u16* __restrict__ Q, const u16* __restrict__ Kb,
                         const u16* __restrict__ Vt, float* __restrict__ Out) {
  const int S = 2048, DM = 2048, DKV = 512;
  __shared__ __align__(32) u16 kbuf[2][32][64];      // [buf][key][dim]  4KB each
  __shared__ __align__(32) u16 vbuf[2][64][32];      // [buf][dim][key]  4KB each
  __shared__ __align__(32) u16 pbuf[8][16][32];      // per-wave P tile

  const int b    = blockIdx.z;
  const int h    = blockIdx.y;
  const int g    = h >> 2;
  const int gs   = h & 3;
  const int tid  = threadIdx.x;
  const int wave = tid >> 5;
  const int lane = tid & 31;
  const int r  = lane & 15;
  const int hi = lane >> 4;
  const int s0 = blockIdx.x * 128 + wave * 16;

  const u16* qrow = Q + ((size_t)b * S + (s0 + r)) * DM + h * 64;
  Frag16 qa[2];
  load_a(qa[0], qrow, 0, hi);
  load_a(qa[1], qrow, 32, hi);

  const u16* kbase = Kb + (size_t)b * S * DKV + g * 64;
  const u16* vbase = Vt + ((size_t)b * 8 + g) * 64 * (size_t)S;

  // per-thread 16B staging slices
  const u16* ksrc0 = kbase + (size_t)(tid >> 3) * DKV + (tid & 7) * 8;  // key-row, dim-col
  const u16* vsrc0 = vbase + (size_t)(tid >> 2) * S + (tid & 3) * 8;    // dim-row, key-col
  const unsigned kl0 = (unsigned)(size_t)(&kbuf[0][0][0]) + tid * 16u;
  const unsigned vl0 = (unsigned)(size_t)(&vbuf[0][0][0]) + tid * 16u;
  const u16* kb0 = &kbuf[0][0][0];
  const u16* vb0 = &vbuf[0][0][0];

  async_16B(ksrc0, kl0);                             // prologue: t=0 -> buf0
  async_16B(vsrc0, vl0);

  v8f acc[4] = {};
  float mrow[8], lrow[8];
#pragma unroll
  for (int j = 0; j < 8; ++j) { mrow[j] = -1e30f; lrow[j] = 0.f; }
  const float scale = 0.125f;                        // 1/sqrt(64)

#pragma unroll 1
  for (int it = 0; it < S / 32; ++it) {
    const int t = it * 32;
    const unsigned par = (unsigned)it & 1u;
    __syncthreads();                                 // prev readers of other buf done
    const int tn = (t + 32 < S) ? t + 32 : 0;        // uniform dummy on last iter
    async_16B(ksrc0 + (size_t)tn * DKV, kl0 + ((par ^ 1u) << 12));
    async_16B(vsrc0 + tn,               vl0 + ((par ^ 1u) << 12));
    asm volatile("s_wait_asynccnt 0x2" ::: "memory"); // current buf's 2 copies landed
    __syncthreads();                                 // all waves' copies visible

    const u16* kt = kb0 + (par << 11);               // 2048 u16 per buffer
    const u16* vt = vb0 + (par << 11);

    // scores: two 16-key column tiles, d_k=64 = 2 WMMA k-steps each
    v8f sf[2] = {};
#pragma unroll
    for (int tf = 0; tf < 2; ++tf) {
      const u16* krow = kt + (tf * 16 + r) * 64;
#pragma unroll
      for (int kk = 0; kk < 2; ++kk) {
        Frag16 kf; load_b(kf, krow, kk * 32, hi);
        sf[tf] = __builtin_amdgcn_wmma_f32_16x16x32_bf16(
            false, qa[kk].v, false, kf.v, (short)0, sf[tf], false, false);
      }
    }
    // online softmax; row j+8*hi lives in this 16-lane half-group
#pragma unroll
    for (int j = 0; j < 8; ++j) {
      float v0 = sf[0][j] * scale, v1 = sf[1][j] * scale;
      float tm = rmax16(fmaxf(v0, v1));
      float mnew = fmaxf(mrow[j], tm);
      float p0 = __expf(v0 - mnew);
      float p1 = __expf(v1 - mnew);
      float corr = __expf(mrow[j] - mnew);
      float ps = rsum16(p0 + p1);
      lrow[j] = lrow[j] * corr + ps;
      mrow[j] = mnew;
#pragma unroll
      for (int nt = 0; nt < 4; ++nt) acc[nt][j] *= corr;
      pbuf[wave][j + 8 * hi][r]      = f2bf(p0);
      pbuf[wave][j + 8 * hi][r + 16] = f2bf(p1);
    }
    asm volatile("s_wait_dscnt 0" ::: "memory");     // wave-local LDS RAW fence
    Frag16 pa;                                       // re-fragment P as WMMA A
    pa.h[0] = *(const v8bf*)(&pbuf[wave][r][hi * 8]);
    pa.h[1] = *(const v8bf*)(&pbuf[wave][r][16 + hi * 8]);
#pragma unroll
    for (int nt = 0; nt < 4; ++nt) {
      Frag16 vf;
      load_b(vf, vt + (nt * 16 + r) * 32, 0, hi);
      acc[nt] = __builtin_amdgcn_wmma_f32_16x16x32_bf16(
          false, pa.v, false, vf.v, (short)0, acc[nt], false, false);
    }
  }

  // normalize + store in the reference's (faithfully scrambled) layout:
  // flat[b][ ((gs*8+g)*S + s)*64 + d ]
  float invl[8];
#pragma unroll
  for (int j = 0; j < 8; ++j) invl[j] = 1.f / lrow[j];
  float* obase = Out + (size_t)b * S * DM + (size_t)(gs * 8 + g) * S * 64;
#pragma unroll
  for (int nt = 0; nt < 4; ++nt)
#pragma unroll
    for (int j = 0; j < 8; ++j) {
      int m = j + 8 * hi;
      obase[(size_t)(s0 + m) * 64 + nt * 16 + r] = acc[nt][j] * invl[j];
    }
}

extern "C" void kernel_launch(void* const* d_in, const int* in_sizes, int n_in,
                              void* d_out, int out_size, void* d_ws, size_t ws_size,
                              hipStream_t stream) {
  const float* x  = (const float*)d_in[0];
  const float* Wq = (const float*)d_in[1];
  const float* bq = (const float*)d_in[2];
  const float* Wk = (const float*)d_in[3];
  const float* bk = (const float*)d_in[4];
  const float* Wv = (const float*)d_in[5];
  const float* bv = (const float*)d_in[6];
  float* out = (float*)d_out;

  const int B = 2, S = 2048, D = 2048, NKV = 512;

  char* ws = (char*)d_ws;
  u16* xb   = (u16*)ws; ws += (size_t)B * S * D * 2;     // 16 MB
  u16* wqt  = (u16*)ws; ws += (size_t)D * D * 2;         //  8 MB
  u16* wkt  = (u16*)ws; ws += (size_t)NKV * D * 2;       //  2 MB
  u16* wvt  = (u16*)ws; ws += (size_t)NKV * D * 2;       //  2 MB
  u16* Qb   = (u16*)ws; ws += (size_t)B * S * D * 2;     // 16 MB
  u16* Kbuf = (u16*)ws; ws += (size_t)B * S * NKV * 2;   //  4 MB
  u16* Vbuf = (u16*)ws; ws += (size_t)B * S * NKV * 2;   //  4 MB
  u16* Vt   = (u16*)ws; ws += (size_t)B * S * NKV * 2;   //  4 MB  (total ~56 MB)

  cast_bf<<<(B * S * D + 255) / 256, 256, 0, stream>>>(x, xb, B * S * D);
  transpose_w<<<(D * D + 255) / 256, 256, 0, stream>>>(Wq, wqt, D, D);
  transpose_w<<<(D * NKV + 255) / 256, 256, 0, stream>>>(Wk, wkt, D, NKV);
  transpose_w<<<(D * NKV + 255) / 256, 256, 0, stream>>>(Wv, wvt, D, NKV);

  proj_gemm<<<dim3(D / 64,   S / 128, B), 256, 0, stream>>>(xb, wqt, bq, Qb,   D);
  proj_gemm<<<dim3(NKV / 64, S / 128, B), 256, 0, stream>>>(xb, wkt, bk, Kbuf, NKV);
  proj_gemm<<<dim3(NKV / 64, S / 128, B), 256, 0, stream>>>(xb, wvt, bv, Vbuf, NKV);

  transpose_v<<<(B * 8 * 64 * S + 255) / 256, 256, 0, stream>>>(Vbuf, Vt);

  gqa_attn<<<dim3(S / 128, 32, B), 256, 0, stream>>>(Qb, Kbuf, Vt, out);
}
